// newLocalGlobalInfoNCE_56444460204190
// MI455X (gfx1250) — matrix operand order repaired and checked by reading, not compiled, not run on hardware
//
#include <hip/hip_runtime.h>

// ---------------- types ----------------
typedef __attribute__((ext_vector_type(16))) __bf16         v16bf;
typedef __attribute__((ext_vector_type(16))) unsigned short v16u16;
typedef __attribute__((ext_vector_type(8)))  float          v8f;
typedef __attribute__((ext_vector_type(4)))  float          v4f;
typedef __attribute__((ext_vector_type(4)))  unsigned int   v4u;

#define NCLASS 64
#define DDIM   512
#define NROWS  65536
#define KSIM   16

// workspace layout (floats)
#define OFF_SUMS   0            // 64*512 = 32768 floats
#define OFF_COUNTS 32768        // 64 floats
#define OFF_ACC    32832        // 2 floats (local, global)
#define OFF_CENT_F 32836        // bf16 centroids start here (64*512 ushorts = 64KB)

static __device__ inline v8f wmma_bf16(v16bf a, v16bf b, v8f c) {
  return __builtin_amdgcn_wmma_f32_16x16x32_bf16(
      /*neg_a=*/false, a, /*neg_b=*/false, b,
      /*c_mod=*/(short)0, c, /*reuse_a=*/false, /*reuse_b=*/false);
}

// A fragment: lane holds row m, K = col0..col0+15 (elems 0..15)
// Use native bf16 converts (v_cvt_pk_bf16_f32) instead of manual bit twiddling.
static __device__ inline v16bf load_a_frag(const float* __restrict__ row, int col0) {
  v4f x0 = *reinterpret_cast<const v4f*>(row + col0);
  v4f x1 = *reinterpret_cast<const v4f*>(row + col0 + 4);
  v4f x2 = *reinterpret_cast<const v4f*>(row + col0 + 8);
  v4f x3 = *reinterpret_cast<const v4f*>(row + col0 + 12);
  v16bf r;
#pragma unroll
  for (int e = 0; e < 4; ++e) {
    r[e]      = (__bf16)x0[e];
    r[4 + e]  = (__bf16)x1[e];
    r[8 + e]  = (__bf16)x2[e];
    r[12 + e] = (__bf16)x3[e];
  }
  return r;
}

// B fragment: lane holds class n, 16 consecutive K values starting at col0 (32B aligned)
static __device__ inline v16bf load_b_frag(const unsigned short* __restrict__ lds,
                                           int n, int col0) {
  const v16u16* p = reinterpret_cast<const v16u16*>(lds + n * DDIM + col0);
  return __builtin_bit_cast(v16bf, *p);
}

// half-wave (16-lane) CE for one row: x0..x3 are logits for classes {lane%16 + 16*g}
static __device__ inline float row_ce(float x0, float x1, float x2, float x3,
                                      int lab, int lane) {
  float mx = fmaxf(fmaxf(x0, x1), fmaxf(x2, x3));
  mx = fmaxf(mx, __shfl_xor(mx, 1));
  mx = fmaxf(mx, __shfl_xor(mx, 2));
  mx = fmaxf(mx, __shfl_xor(mx, 4));
  mx = fmaxf(mx, __shfl_xor(mx, 8));
  float s = expf(x0 - mx) + expf(x1 - mx) + expf(x2 - mx) + expf(x3 - mx);
  s += __shfl_xor(s, 1);
  s += __shfl_xor(s, 2);
  s += __shfl_xor(s, 4);
  s += __shfl_xor(s, 8);
  const int g = lab >> 4;
  float xs = (g == 0) ? x0 : (g == 1) ? x1 : (g == 2) ? x2 : x3;
  float cont = ((lane & 15) == (lab & 15)) ? xs : 0.0f;
  cont += __shfl_xor(cont, 1);
  cont += __shfl_xor(cont, 2);
  cont += __shfl_xor(cont, 4);
  cont += __shfl_xor(cont, 8);
  return mx + logf(s) - cont;   // = -(x_lab - mx - log(sum exp))
}

// ---------------- kernel 1: zero workspace ----------------
__global__ void zero_ws(float* __restrict__ ws, int n) {
  int i = blockIdx.x * blockDim.x + threadIdx.x;
  if (i < n) ws[i] = 0.0f;
}

// ---------------- kernel 2: centroid partial sums ----------------
// grid (8 column slices, 32 row chunks); block (64, 4)
__global__ __launch_bounds__(256) void centroid_partial(
    const float* __restrict__ S1, const int* __restrict__ labels,
    float* __restrict__ sums, float* __restrict__ counts) {
  __shared__ float lds[NCLASS * 64];
  __shared__ float cnt[NCLASS];
  const int x = threadIdx.x;                 // column within slice
  const int y = threadIdx.y;
  const int tid = y * 64 + x;
  for (int i = tid; i < NCLASS * 64; i += 256) lds[i] = 0.0f;
  if (tid < NCLASS) cnt[tid] = 0.0f;
  __syncthreads();

  const int colBase = blockIdx.x << 6;
  const int rStart  = blockIdx.y * 2048;
  for (int r = rStart + y; r < rStart + 2048; r += 4) {
    const int lab = labels[r];
    const float v = S1[(size_t)r * DDIM + colBase + x];
    atomicAdd(&lds[lab * 64 + x], v);
    if (blockIdx.x == 0 && x == 0) atomicAdd(&cnt[lab], 1.0f);
  }
  __syncthreads();

  for (int i = tid; i < NCLASS * 64; i += 256) {
    const int c = i >> 6, d = i & 63;
    atomicAdd(&sums[c * DDIM + colBase + d], lds[i]);
  }
  if (blockIdx.x == 0 && tid < NCLASS) atomicAdd(&counts[tid], cnt[tid]);
}

// ---------------- kernel 3: centroids -> bf16, fold 1/TEMP ----------------
__global__ void finalize_centroids(const float* __restrict__ sums,
                                   const float* __restrict__ counts,
                                   unsigned short* __restrict__ cent) {
  int i = blockIdx.x * blockDim.x + threadIdx.x;
  if (i >= NCLASS * DDIM) return;
  const int c = i >> 9;  // /512
  const float v = (sums[i] / counts[c]) * 10.0f;  // fold 1/TEMP into B
  cent[i] = __builtin_bit_cast(unsigned short, (__bf16)v);
}

// ---------------- kernel 4: fused dual GEMM + weighted CE ----------------
// 512 blocks x 256 threads; 8 waves/block, one 16-row tile per wave; 64KB dyn LDS
__global__ __launch_bounds__(256) void fused_logits_ce(
    const float* __restrict__ S1, const float* __restrict__ S2,
    const int* __restrict__ labels, const float* __restrict__ sim,
    const unsigned short* __restrict__ centBf, float* __restrict__ accum) {
  extern __shared__ unsigned short ldsC[];   // [64][512] bf16 = 64KB
  const int tid = threadIdx.x;
  {
    v4u* dst = reinterpret_cast<v4u*>(ldsC);
    const v4u* src = reinterpret_cast<const v4u*>(centBf);
#pragma unroll
    for (int i = 0; i < 16; ++i) dst[tid + 256 * i] = src[tid + 256 * i];
  }
  __syncthreads();

  const int wave = tid >> 5, lane = tid & 31;
  const int m = lane & 15, kh = lane >> 4;
  const int tile = blockIdx.x * 8 + wave;
  const int r0 = tile << 4;

  // per-lane label + CE weight for row r0 + (lane&15)
  int labv; float wv;
  {
    const int rr = r0 + m;
    labv = labels[rr];
    const v4f* sp = reinterpret_cast<const v4f*>(sim + (size_t)rr * KSIM);
    v4f a = sp[0], b = sp[1], c = sp[2], d = sp[3];
    float t = 0.0f;
#pragma unroll
    for (int e = 0; e < 4; ++e) t += a[e] + b[e] + c[e] + d[e];
    wv = t * (1.0f / 16.0f);
  }

  const v8f vzero = {0.f, 0.f, 0.f, 0.f, 0.f, 0.f, 0.f, 0.f};
  v8f accL[4], accG[4];
#pragma unroll
  for (int g = 0; g < 4; ++g) { accL[g] = vzero; accG[g] = vzero; }

  const float* rowA = S1 + (size_t)(r0 + m) * DDIM;
  const float* rowB = S2 + (size_t)(r0 + m) * DDIM;

#pragma unroll 2
  for (int ks = 0; ks < 16; ++ks) {
    const int kb = ks * 32;
    v16bf a1 = load_a_frag(rowA, kb + 16 * kh);
    v16bf a2 = load_a_frag(rowB, kb + 16 * kh);
    const int bcol = kb + 16 * kh;
#pragma unroll
    for (int g = 0; g < 4; ++g) {
      v16bf bf = load_b_frag(ldsC, (g << 4) + m, bcol);
      accL[g] = wmma_bf16(a1, bf, accL[g]);
      accG[g] = wmma_bf16(a2, bf, accG[g]);
    }
  }

  // per-row log-softmax CE; VGPR slot j holds row j (lanes 0-15) / row 8+j (lanes 16-31)
  float sumL = 0.0f, sumG = 0.0f;
#pragma unroll
  for (int j = 0; j < 8; ++j) {
    const int src = j + 8 * kh;                 // lane holding this row's label/weight
    const int lab = __shfl(labv, src, 32);
    const float w = __shfl(wv, src, 32);
    const float ceL = row_ce(accL[0][j], accL[1][j], accL[2][j], accL[3][j], lab, lane);
    const float ceG = row_ce(accG[0][j], accG[1][j], accG[2][j], accG[3][j], lab, lane);
    sumL += ceL * w;
    sumG += ceG * w;
  }
  if (m == 0) {                                  // lanes 0 and 16: one per half-wave
    atomicAdd(&accum[0], sumL);
    atomicAdd(&accum[1], sumG);
  }
}

// ---------------- kernel 5: final scalar combine ----------------
__global__ void final_combine(const float* __restrict__ acc, float* __restrict__ out) {
  // ((1-GLW)*L + GLW*G)/2 with GLW=0.5 and mean over N  ->  (sumL+sumG)/(4N)
  out[0] = (acc[0] + acc[1]) * (1.0f / (4.0f * (float)NROWS));
}

extern "C" void kernel_launch(void* const* d_in, const int* in_sizes, int n_in,
                              void* d_out, int out_size, void* d_ws, size_t ws_size,
                              hipStream_t stream) {
  const float* S1  = (const float*)d_in[0];
  const float* S2  = (const float*)d_in[1];
  const int*   seg = (const int*)d_in[2];
  const float* sim = (const float*)d_in[3];
  float* out = (float*)d_out;

  float* ws     = (float*)d_ws;
  float* sums   = ws + OFF_SUMS;
  float* counts = ws + OFF_COUNTS;
  float* accum  = ws + OFF_ACC;
  unsigned short* cent = (unsigned short*)(ws + OFF_CENT_F);

  zero_ws<<<(OFF_CENT_F + 255) / 256, 256, 0, stream>>>(ws, OFF_CENT_F);
  centroid_partial<<<dim3(8, 32), dim3(64, 4), 0, stream>>>(S1, seg, sums, counts);
  finalize_centroids<<<(NCLASS * DDIM + 255) / 256, 256, 0, stream>>>(sums, counts, cent);
  fused_logits_ce<<<NROWS / 16 / 8, 256, NCLASS * DDIM * sizeof(unsigned short), stream>>>(
      S1, S2, seg, sim, cent, accum);
  final_combine<<<1, 1, 0, stream>>>(accum, out);
}
